// BiGCN_graphcl_20194936226505
// MI455X (gfx1250) — matrix operand order repaired and checked by reading, not compiled, not run on hardware
//
#include <hip/hip_runtime.h>
#include <hip/hip_bf16.h>
#include <stdint.h>
#include <stddef.h>

// ---------------------------------------------------------------------------
// BiGCN (graphcl) fused pipeline for MI455X (gfx1250, wave32, WMMA bf16)
//   N=100000 nodes, E=100000 edges, B=128 graphs, IN=256, HID=OUT=128
// Dense transforms via v_wmma_f32_16x16x32_bf16; edge scatter via
// global_atomic_add_f32 (L2-resident: Hs+AGG = ~100MB < 192MB L2).
// ---------------------------------------------------------------------------

typedef __bf16 bf16;
typedef __attribute__((ext_vector_type(16))) bf16  v16bf;
typedef __attribute__((ext_vector_type(8)))  bf16  v8bf;
typedef __attribute__((ext_vector_type(8)))  float v8f;
typedef __attribute__((ext_vector_type(4)))  unsigned int u32x4;

#define FEAT 128      // HID == OUT
#define INF  256      // IN
#define NGR  128      // num_graphs (python scalar, fixed by reference)

__device__ __forceinline__ bf16 f2bf(float f) {
  unsigned u = __builtin_bit_cast(unsigned, f);
  unsigned r = (u + 0x7FFFu + ((u >> 16) & 1u)) >> 16;   // round-nearest-even
  return __builtin_bit_cast(bf16, (unsigned short)r);
}

// --- roots[g] = searchsorted(batch, g); roots[B] = N -----------------------
__global__ void roots_kernel(const int* __restrict__ batch, int* __restrict__ roots,
                             int N, int B) {
  int g = blockIdx.x * blockDim.x + threadIdx.x;
  if (g > B) return;
  if (g == B) { roots[B] = N; return; }
  int lo = 0, hi = N;
  while (lo < hi) { int mid = (lo + hi) >> 1; if (batch[mid] < g) lo = mid + 1; else hi = mid; }
  roots[g] = lo;
}

// --- pre-swizzle W (K x 128 f32 row-major) into WMMA B-fragment layout -----
// frag f = kt*8+nt ; lane l ; value q(0..15): K = kt*32 + (l>>4)*16 + q,
//                                             Ncol = nt*16 + (l&15)
__global__ void swz_kernel(const float* __restrict__ W, bf16* __restrict__ out, int K) {
  int idx = blockIdx.x * blockDim.x + threadIdx.x;
  if (idx >= K * FEAT) return;
  int q  = idx & 15;
  int l  = (idx >> 4) & 31;
  int f  = idx >> 9;
  int nt = f & 7, kt = f >> 3;
  int krow = kt * 32 + ((l >> 4) << 4) + q;
  int col  = nt * 16 + (l & 15);
  out[idx] = f2bf(W[(size_t)krow * FEAT + col]);
}

// --- xv -> bf16, rows [0,N)=x, row N=emb_w, rows [N+1,Mpad)=0 --------------
__global__ void cvt_kernel(const float* __restrict__ x, const float* __restrict__ emb,
                           bf16* __restrict__ out, int N, int Mpad) {
  size_t idx = (size_t)blockIdx.x * blockDim.x + threadIdx.x;
  size_t total = (size_t)Mpad * INF;
  if (idx >= total) return;
  int row = (int)(idx >> 8), col = (int)(idx & 255);
  float v = 0.0f;
  if (row < N)       v = x[(size_t)row * INF + col];
  else if (row == N) v = emb[col];
  out[idx] = f2bf(v);
}

// --- degree kernels --------------------------------------------------------
__global__ void deg_init(float* __restrict__ deg2, int nv) {
  int i = blockIdx.x * blockDim.x + threadIdx.x;
  if (i < 2 * nv) deg2[i] = 1.0f;                 // self-loop
}
__global__ void deg_edges(const int* __restrict__ esrc, const int* __restrict__ edst,
                          const int* __restrict__ roots,
                          float* __restrict__ deg_td, float* __restrict__ deg_bu,
                          int E, int B, int n) {
  int i = blockIdx.x * blockDim.x + threadIdx.x;
  if (i < E) {
    atomicAdd(&deg_td[edst[i]], 1.0f);
    atomicAdd(&deg_bu[esrc[i]], 1.0f);
  } else if (i < E + B) {
    int g = i - E;
    atomicAdd(&deg_td[roots[g]], 1.0f);
    atomicAdd(&deg_bu[n], 1.0f);
  }
}
__global__ void deg_rsqrt(float* __restrict__ d, int count) {
  int i = blockIdx.x * blockDim.x + threadIdx.x;
  if (i < count) d[i] = rsqrtf(d[i]);
}

__global__ void zero_f32(float* __restrict__ p, size_t count) {
  size_t i = (size_t)blockIdx.x * blockDim.x + threadIdx.x;
  if (i < count) p[i] = 0.0f;
}

// --- WMMA GEMM: Hs(Mpad x 128, f32) = (A(Mpad x K, bf16) @ Wswz) * dinv[row]
// block = 4 waves, each wave owns a 16-row strip, full N=128 in 8 accumulators
__launch_bounds__(128)
__global__ void gemm_kernel(const bf16* __restrict__ A, const bf16* __restrict__ Wswz,
                            const float* __restrict__ dinv, float* __restrict__ Hs,
                            int K, int nv) {
  __shared__ __align__(16) bf16 ldsW[INF * FEAT];   // 64KB max (K=256)
  const int tid  = threadIdx.x;
  const int lane = tid & 31;
  const int wave = tid >> 5;
  const int KT = K >> 5;

  // cooperative staged copy of swizzled weights into LDS (b128 chunks)
  {
    const u32x4* src = (const u32x4*)Wswz;
    u32x4* dst = (u32x4*)ldsW;
    const int chunks = (K * FEAT) >> 3;            // 16B chunks
    for (int i = tid; i < chunks; i += blockDim.x) dst[i] = src[i];
  }
  __syncthreads();

  const int m0    = (blockIdx.x * 4 + wave) * 16;
  const int arow_ = m0 + (lane & 15);
  const int khalf = lane >> 4;
  const bf16* arow = A + (size_t)arow_ * K + (khalf << 4);

  v8f c[8];
  #pragma unroll
  for (int nt = 0; nt < 8; ++nt) c[nt] = (v8f)(0.0f);

  union V16 { v16bf v; v8bf h[2]; };

  for (int kt = 0; kt < KT; ++kt) {
    const int kb = kt << 5;
    V16 a;
    a.h[0] = *(const v8bf*)(arow + kb);            // K = kb + khalf*16 + 0..7
    a.h[1] = *(const v8bf*)(arow + kb + 8);        // K = kb + khalf*16 + 8..15
    const bf16* bbase = ldsW + ((size_t)(kt * 8) * 32 + lane) * 16;
    #pragma unroll
    for (int nt = 0; nt < 8; ++nt) {
      const bf16* bp = bbase + (size_t)nt * 32 * 16;
      V16 b;
      b.h[0] = *(const v8bf*)(bp);
      b.h[1] = *(const v8bf*)(bp + 8);
      c[nt] = __builtin_amdgcn_wmma_f32_16x16x32_bf16(
          false, a.v, false, b.v, (short)0, c[nt], false, false);
    }
  }

  // epilogue: C layout — lane<16: rows m0+0..7, lane>=16: rows m0+8..15; col=lane%16
  const int col   = lane & 15;
  const int rbase = m0 + ((lane >> 4) << 3);
  float dv8[8];
  #pragma unroll
  for (int v = 0; v < 8; ++v) {
    int row = rbase + v;
    dv8[v] = (row < nv) ? dinv[row] : 0.0f;
  }
  #pragma unroll
  for (int nt = 0; nt < 8; ++nt) {
    #pragma unroll
    for (int v = 0; v < 8; ++v) {
      Hs[(size_t)(rbase + v) * FEAT + nt * 16 + col] = c[nt][v] * dv8[v];
    }
  }
}

// --- edge scatter: AGG[d] += Hs[s] (128 feats per edge; +B virtual edges) --
__global__ void scatter_kernel(const float* __restrict__ Hs, float* __restrict__ AGG,
                               const int* __restrict__ esrc, const int* __restrict__ edst,
                               const int* __restrict__ roots,
                               int E, int B, int n, int flip) {
  size_t idx = (size_t)blockIdx.x * blockDim.x + threadIdx.x;
  if (idx >= (size_t)(E + B) * FEAT) return;
  int e = (int)(idx >> 7);
  int c = (int)(idx & (FEAT - 1));
  int s, d;
  if (e < E) { s = esrc[e]; d = edst[e]; }
  else       { s = n; d = roots[e - E]; }          // td virtual: n -> root
  if (flip)  { int t = s; s = d; d = t; }          // bu: flipped edges
  atomicAdd(&AGG[(size_t)d * FEAT + c], Hs[(size_t)s * FEAT + c]);
}

// --- layer-1 finalize: H1b = bf16(relu(dinv*(Hs+AGG)+b1)); zero pad rows ---
__global__ void finalize1(const float* __restrict__ Hs, const float* __restrict__ AGG,
                          const float* __restrict__ dinv, const float* __restrict__ b1,
                          bf16* __restrict__ H1b, int nv, int Mpad) {
  size_t idx = (size_t)blockIdx.x * blockDim.x + threadIdx.x;
  if (idx >= (size_t)Mpad * FEAT) return;
  int row = (int)(idx >> 7), c = (int)(idx & (FEAT - 1));
  float v = 0.0f;
  if (row < nv) {
    v = dinv[row] * (Hs[idx] + AGG[idx]) + b1[c];
    v = fmaxf(v, 0.0f);
  }
  H1b[idx] = f2bf(v);
}

// --- layer-2 finalize in place (no relu); capture virtual-node row ---------
__global__ void finalize2(const float* __restrict__ Hs, float* __restrict__ AGG,
                          const float* __restrict__ dinv, const float* __restrict__ b2,
                          float* __restrict__ vnode, int nv, int n) {
  size_t idx = (size_t)blockIdx.x * blockDim.x + threadIdx.x;
  if (idx >= (size_t)nv * FEAT) return;
  int row = (int)(idx >> 7), c = (int)(idx & (FEAT - 1));
  float v = dinv[row] * (Hs[idx] + AGG[idx]) + b2[c];
  AGG[idx] = v;
  if (row == n) vnode[c] = v;
}

// --- per-graph pooling: sum rows [roots[g], roots[g+1]) + virtual row ------
__global__ void pool_kernel(const float* __restrict__ H2, const int* __restrict__ roots,
                            const float* __restrict__ vnode, float* __restrict__ pooled) {
  int g = blockIdx.x, c = threadIdx.x;            // 128 threads
  int s = roots[g], e = roots[g + 1];
  float acc = vnode[c];
  for (int r = s; r < e; ++r) acc += H2[(size_t)r * FEAT + c];
  pooled[(size_t)g * FEAT + c] = acc;
}

// --- final MLP: out = relu(concat(bu,td) @ p_w1 + p_b1) @ p_w2 + p_b2 ------
__global__ void mlp_kernel(const float* __restrict__ pooled_td, const float* __restrict__ pooled_bu,
                           const float* __restrict__ w1, const float* __restrict__ b1,
                           const float* __restrict__ w2, const float* __restrict__ b2,
                           float* __restrict__ out) {
  __shared__ float h[256];
  __shared__ float t[256];
  int g = blockIdx.x, tid = threadIdx.x;          // 256 threads
  h[tid] = (tid < 128) ? pooled_bu[(size_t)g * 128 + tid]
                       : pooled_td[(size_t)g * 128 + (tid - 128)];
  __syncthreads();
  float acc = b1[tid];
  for (int k = 0; k < 256; ++k) acc = fmaf(h[k], w1[(size_t)k * 256 + tid], acc);
  t[tid] = fmaxf(acc, 0.0f);
  __syncthreads();
  if (tid < 128) {
    float o = b2[tid];
    for (int k = 0; k < 256; ++k) o = fmaf(t[k], w2[(size_t)k * 128 + tid], o);
    out[(size_t)g * 128 + tid] = o;
  }
}

// ---------------------------------------------------------------------------
static inline int cdiv(size_t a, int b) { return (int)((a + b - 1) / b); }

extern "C" void kernel_launch(void* const* d_in, const int* in_sizes, int n_in,
                              void* d_out, int out_size, void* d_ws, size_t ws_size,
                              hipStream_t stream) {
  (void)n_in; (void)out_size; (void)ws_size;
  const float* x     = (const float*)d_in[0];
  const float* emb_w = (const float*)d_in[1];
  const float* td_w1 = (const float*)d_in[2];
  const float* td_b1 = (const float*)d_in[3];
  const float* td_w2 = (const float*)d_in[4];
  const float* td_b2 = (const float*)d_in[5];
  const float* bu_w1 = (const float*)d_in[6];
  const float* bu_b1 = (const float*)d_in[7];
  const float* bu_w2 = (const float*)d_in[8];
  const float* bu_b2 = (const float*)d_in[9];
  const float* p_w1  = (const float*)d_in[10];
  const float* p_b1  = (const float*)d_in[11];
  const float* p_w2  = (const float*)d_in[12];
  const float* p_b2  = (const float*)d_in[13];
  const int*   eidx  = (const int*)d_in[14];
  const int*   batch = (const int*)d_in[15];

  const int N  = in_sizes[0] / INF;
  const int E  = in_sizes[14] / 2;
  const int B  = NGR;
  const int nv = N + 1;
  const int Mpad = ((nv + 63) / 64) * 64;
  const int* esrc = eidx;
  const int* edst = eidx + E;

  // ---- workspace carve-out (256B aligned) ----
  char* w = (char*)d_ws;
  auto carve = [&](size_t bytes) -> void* {
    void* p = (void*)w; w += (bytes + 255) & ~(size_t)255; return p;
  };
  bf16*  xvb     = (bf16*)carve((size_t)Mpad * INF * sizeof(bf16));
  bf16*  swz_td1 = (bf16*)carve((size_t)INF * FEAT * sizeof(bf16));
  bf16*  swz_bu1 = (bf16*)carve((size_t)INF * FEAT * sizeof(bf16));
  bf16*  swz_td2 = (bf16*)carve((size_t)FEAT * FEAT * sizeof(bf16));
  bf16*  swz_bu2 = (bf16*)carve((size_t)FEAT * FEAT * sizeof(bf16));
  float* dinv2   = (float*)carve((size_t)2 * nv * sizeof(float)); // [td | bu]
  int*   roots   = (int*)carve((size_t)(B + 1) * sizeof(int));
  float* Hs      = (float*)carve((size_t)Mpad * FEAT * sizeof(float));
  float* AGG     = (float*)carve((size_t)Mpad * FEAT * sizeof(float));
  bf16*  H1b     = (bf16*)carve((size_t)Mpad * FEAT * sizeof(bf16));
  float* pooled  = (float*)carve((size_t)2 * B * FEAT * sizeof(float));
  float* vnode   = (float*)carve((size_t)2 * FEAT * sizeof(float));

  // ---- prep ----
  roots_kernel<<<1, 256, 0, stream>>>(batch, roots, N, B);
  swz_kernel<<<cdiv((size_t)INF * FEAT, 256), 256, 0, stream>>>(td_w1, swz_td1, INF);
  swz_kernel<<<cdiv((size_t)INF * FEAT, 256), 256, 0, stream>>>(bu_w1, swz_bu1, INF);
  swz_kernel<<<cdiv((size_t)FEAT * FEAT, 256), 256, 0, stream>>>(td_w2, swz_td2, FEAT);
  swz_kernel<<<cdiv((size_t)FEAT * FEAT, 256), 256, 0, stream>>>(bu_w2, swz_bu2, FEAT);
  cvt_kernel<<<cdiv((size_t)Mpad * INF, 256), 256, 0, stream>>>(x, emb_w, xvb, N, Mpad);

  deg_init<<<cdiv((size_t)2 * nv, 256), 256, 0, stream>>>(dinv2, nv);
  deg_edges<<<cdiv((size_t)(E + B), 256), 256, 0, stream>>>(esrc, edst, roots,
                                                            dinv2, dinv2 + nv, E, B, N);
  deg_rsqrt<<<cdiv((size_t)2 * nv, 256), 256, 0, stream>>>(dinv2, 2 * nv);

  // ---- two GCN branches (td: flip=0, bu: flip=1) ----
  for (int br = 0; br < 2; ++br) {
    const float* dv = dinv2 + (size_t)br * nv;
    const bf16*  W1 = br ? swz_bu1 : swz_td1;
    const bf16*  W2 = br ? swz_bu2 : swz_td2;
    const float* b1 = br ? bu_b1 : td_b1;
    const float* b2 = br ? bu_b2 : td_b2;

    zero_f32<<<cdiv((size_t)Mpad * FEAT, 256), 256, 0, stream>>>(AGG, (size_t)Mpad * FEAT);
    gemm_kernel<<<Mpad / 64, 128, 0, stream>>>(xvb, W1, dv, Hs, INF, nv);
    scatter_kernel<<<cdiv((size_t)(E + B) * FEAT, 256), 256, 0, stream>>>(
        Hs, AGG, esrc, edst, roots, E, B, N, br);
    finalize1<<<cdiv((size_t)Mpad * FEAT, 256), 256, 0, stream>>>(
        Hs, AGG, dv, b1, H1b, nv, Mpad);

    zero_f32<<<cdiv((size_t)Mpad * FEAT, 256), 256, 0, stream>>>(AGG, (size_t)Mpad * FEAT);
    gemm_kernel<<<Mpad / 64, 128, 0, stream>>>(H1b, W2, dv, Hs, FEAT, nv);
    scatter_kernel<<<cdiv((size_t)(E + B) * FEAT, 256), 256, 0, stream>>>(
        Hs, AGG, esrc, edst, roots, E, B, N, br);
    finalize2<<<cdiv((size_t)nv * FEAT, 256), 256, 0, stream>>>(
        Hs, AGG, dv, b2, vnode + (size_t)br * FEAT, nv, N);

    pool_kernel<<<B, FEAT, 0, stream>>>(AGG, roots, vnode + (size_t)br * FEAT,
                                        pooled + (size_t)br * B * FEAT);
  }

  // ---- final MLP: h = concat(bu, td) ----
  mlp_kernel<<<B, 256, 0, stream>>>(pooled, pooled + (size_t)B * FEAT,
                                    p_w1, p_b1, p_w2, p_b2, (float*)d_out);
}
